// NAM_4818953306387
// MI455X (gfx1250) — matrix operand order repaired and checked by reading, not compiled, not run on hardware
//
#include <hip/hip_runtime.h>

typedef __attribute__((ext_vector_type(16))) _Float16 v16h;
typedef __attribute__((ext_vector_type(8)))  float    v8f;

#define B_   16384
#define F_   64
#define U_   64
#define WAVES 4
#define TPB  (WAVES * 32)

// ---------------------------------------------------------------------------
// Kernel 1: pack exp(exu_w) into the WMMA A-matrix K-order (f32), convert
// lin_w to f16 (natural order == B-matrix K-order), and reduce sum(lin_b).
// A 16-bit 16x32 layout (contiguous K, matching B): lanes 0-15 hold K 0..15
// (u=e); lanes 16-31 hold K 16..31 (u=e+16). Inverse map below. Chunk c
// covers u in [32c, 32c+32).
// ---------------------------------------------------------------------------
__global__ void nam_pack_kernel(const float* __restrict__ exu_w,
                                const float* __restrict__ lin_w,
                                const float* __restrict__ lin_b,
                                float*    __restrict__ ws_ew,   // [F][2][2][16]
                                _Float16* __restrict__ ws_bf,   // [F][64]
                                float*    __restrict__ ws_sum)  // [1]
{
    int idx = blockIdx.x * blockDim.x + threadIdx.x;   // 0..4095
    if (idx < F_ * U_) {
        int f = idx >> 6, u = idx & 63;
        int c = u >> 5, r = u & 31;
        int hi, e;
        hi = r >> 4; e = r & 15;
        ws_ew[((f * 2 + c) * 2 + hi) * 16 + e] = expf(exu_w[idx]);
        ws_bf[idx] = (_Float16)lin_w[idx];
        if (idx == 0) {
            float s = 0.f;
            for (int i = 0; i < F_; ++i) s += lin_b[i];
            ws_sum[0] = s;
        }
    }
}

// ---------------------------------------------------------------------------
// Kernel 2: main NAM forward. One wave32 per 16-row batch tile.
// Per feature f: build A (h tile, f16) on the fly, B = lin_w[f,:] broadcast
// into all 16 columns, and chain C across features so the accumulator ends
// as sum_f dot[b,f] (== out[b] before biases). Per-feature fnn values are
// the accumulator deltas.
// ---------------------------------------------------------------------------
__global__ void __launch_bounds__(TPB)
nam_main_kernel(const float*    __restrict__ x,       // [B][F]
                const float*    __restrict__ exu_b,   // [F]
                const float*    __restrict__ lin_b,   // [F]
                const float*    __restrict__ bias,    // [1]
                const float*    __restrict__ ws_ew,   // [F][2][2][16]
                const _Float16* __restrict__ ws_bf,   // [F][64]
                const float*    __restrict__ ws_sum,  // [1]
                float*          __restrict__ out,     // [B]
                float*          __restrict__ fnn)     // [B][F]
{
    __shared__ float    ldsE[F_ * U_];          // 16 KB: exp(w) in A K-order
    __shared__ _Float16 ldsB[F_ * U_];          //  8 KB: lin_w f16, B K-order
    __shared__ float    ldsEb[F_];
    __shared__ float    ldsLb[F_];
    __shared__ float    ldsFnn[WAVES][16][F_];  // 16 KB: per-wave fnn tile

    const int t = threadIdx.x;
    for (int i = t; i < F_ * U_; i += TPB) { ldsE[i] = ws_ew[i]; ldsB[i] = ws_bf[i]; }
    for (int i = t; i < F_;      i += TPB) { ldsEb[i] = exu_b[i]; ldsLb[i] = lin_b[i]; }
    __syncthreads();

    const int wave = t >> 5;
    const int lane = t & 31;
    const int row  = lane & 15;     // batch row within tile this lane owns
    const int hi   = lane >> 4;     // lane half selects K-subset of A/B
    const int base = (blockIdx.x * WAVES + wave) * 16;

    const float sumLb = ws_sum[0] + bias[0];
    const float* xrow = x + (size_t)(base + row) * F_;

    v8f acc = {};   // chained across features -> per-row sum over f

    for (int f = 0; f < F_; ++f) {
        // s = relu(x - b); h[u] = min(s * exp(w[u]), 1)
        const float s = fmaxf(xrow[f] - ldsEb[f], 0.f);
        const float* ew0 = &ldsE[((f * 2 + 0) * 2 + hi) * 16];
        const float* ew1 = &ldsE[((f * 2 + 1) * 2 + hi) * 16];
        v16h a0, a1;
#pragma unroll
        for (int e = 0; e < 16; ++e) {
            a0[e] = (_Float16)fminf(s * ew0[e], 1.0f);
            a1[e] = (_Float16)fminf(s * ew1[e], 1.0f);
        }
        // B: all 16 columns = lin_w[f, k-chunk] (broadcast LDS reads)
        const v16h b0 = *(const v16h*)&ldsB[f * U_ + hi * 16];
        const v16h b1 = *(const v16h*)&ldsB[f * U_ + 32 + hi * 16];

        const v8f prev = acc;
        acc = __builtin_amdgcn_wmma_f32_16x16x32_f16(false, a0, false, b0,
                                                     (short)0, acc, false, false);
        acc = __builtin_amdgcn_wmma_f32_16x16x32_f16(false, a1, false, b1,
                                                     (short)0, acc, false, false);

        // C/D layout: VGPR j = row j (lanes 0-15) / row j+8 (lanes 16-31),
        // identical across columns -> lanes 0 and 16 stage the 16 row deltas.
        if (row == 0) {
            const int rb = hi * 8;
#pragma unroll
            for (int j = 0; j < 8; ++j)
                ldsFnn[wave][rb + j][f] = acc[j] - prev[j];
        }
    }

    // out[b] = sum_f dot + sum(lin_b) + bias  (accumulator already holds it)
    if (row == 0) {
        const int rb = hi * 8;
#pragma unroll
        for (int j = 0; j < 8; ++j)
            out[base + rb + j] = acc[j] + sumLb;
    }
    __syncthreads();

    // Coalesced fnn store: block tile = (WAVES*16) rows x 64 cols
    const int gBase = blockIdx.x * (WAVES * 16);
    for (int i = t; i < WAVES * 16 * F_; i += TPB) {
        const int r  = i >> 6;
        const int fc = i & 63;
        fnn[(size_t)(gBase + r) * F_ + fc] = ldsFnn[r >> 4][r & 15][fc] + ldsLb[fc];
    }
}

// ---------------------------------------------------------------------------
extern "C" void kernel_launch(void* const* d_in, const int* in_sizes, int n_in,
                              void* d_out, int out_size, void* d_ws, size_t ws_size,
                              hipStream_t stream) {
    (void)in_sizes; (void)n_in; (void)out_size; (void)ws_size;
    const float* x     = (const float*)d_in[0];   // [B,F]
    const float* exu_w = (const float*)d_in[1];   // [F,U]
    const float* exu_b = (const float*)d_in[2];   // [F]
    const float* lin_w = (const float*)d_in[3];   // [F,U]
    const float* lin_b = (const float*)d_in[4];   // [F]
    const float* bias  = (const float*)d_in[5];   // [1]

    float* out = (float*)d_out;       // [0, B): out
    float* fnn = out + B_;            // [B, B + B*F): fnn_out

    float*    ws_ew  = (float*)d_ws;                                   // 16 KB
    _Float16* ws_bf  = (_Float16*)((char*)d_ws + F_ * U_ * 4);         //  8 KB
    float*    ws_sum = (float*)((char*)d_ws + F_ * U_ * 4 + F_ * U_ * 2);

    nam_pack_kernel<<<(F_ * U_ + 255) / 256, 256, 0, stream>>>(
        exu_w, lin_w, lin_b, ws_ew, ws_bf, ws_sum);

    nam_main_kernel<<<B_ / (WAVES * 16), TPB, 0, stream>>>(
        x, exu_b, lin_b, bias, ws_ew, ws_bf, ws_sum, out, fnn);
}
